// SelfPromptingHead_60988535603855
// MI455X (gfx1250) — compile-verified, matrix operand order checked by reading, not run on hardware
//
#include <hip/hip_runtime.h>
#include <hip/hip_bf16.h>
#include <math.h>

typedef __attribute__((ext_vector_type(16))) _Float16 v16h;
typedef __attribute__((ext_vector_type(8)))  _Float16 v8h;
typedef __attribute__((ext_vector_type(8)))  float    v8f;
typedef __attribute__((ext_vector_type(4)))  int      v4i;

#define BATCH 32
#define OUTHW 256

// ---- CDNA5 async direct-to-LDS copy (gated: falls back to vector ld/st) ----
#if defined(__has_builtin)
#if __has_builtin(__builtin_amdgcn_global_load_async_to_lds_b128) && \
    __has_builtin(__builtin_amdgcn_s_wait_asynccnt)
#define HAVE_ASYNC_LDS 1
#endif
#endif
#ifndef HAVE_ASYNC_LDS
#define HAVE_ASYNC_LDS 0
#endif

#if HAVE_ASYNC_LDS
typedef __attribute__((address_space(1))) v4i gbl_v4i;
typedef __attribute__((address_space(3))) v4i lds_v4i;
#endif

// ------------------------------------------------------------------
// Repack conv weights [cout][cin][3][3] f32 -> [tap][cin][cout] f16
// ------------------------------------------------------------------
__global__ void repack_weights(const float* __restrict__ w1,
                               const float* __restrict__ w2,
                               _Float16* __restrict__ wb1,
                               _Float16* __restrict__ wb2) {
  int i = blockIdx.x * blockDim.x + threadIdx.x;
  const int N1 = 64 * 256 * 9;
  const int N2 = 64 * 64 * 9;
  if (i < N1) {
    int cout = i / (256 * 9);
    int r    = i % (256 * 9);
    int cin  = r / 9, tap = r % 9;
    wb1[(tap * 256 + cin) * 64 + cout] = (_Float16)w1[i];
  } else if (i < N1 + N2) {
    int j    = i - N1;
    int cout = j / (64 * 9);
    int r    = j % (64 * 9);
    int cin  = r / 9, tap = r % 9;
    wb2[(tap * 64 + cin) * 64 + cout] = (_Float16)w2[j];
  }
}

// ------------------------------------------------------------------
// 3x3 conv (pad 1) + inference-BN + ReLU as implicit GEMM on WMMA.
// Block = 256 thr = 8 waves; tile = 8 rows x 16 cols x 64 cout of one image.
// Wave w computes output row w: 16(px) x 64(cout) via 4x v_wmma_f32_16x16x32_f16
// per tap, all 9 taps of weights staged in LDS -> one barrier pair per K-chunk.
// conv1: input NCHW f32 (converted to f16 while staging).
// conv2: input NHWC f16 (contiguous 16B copies; async-to-LDS when available).
// Output activations are NHWC f16.
// ------------------------------------------------------------------
template <int CIN, typename IT>
__global__ __launch_bounds__(256)
void conv3x3_bn_relu(const IT* __restrict__ in,
                     const _Float16* __restrict__ wb,  // [9][CIN][64]
                     const float* __restrict__ gam, const float* __restrict__ bet,
                     const float* __restrict__ mu,  const float* __restrict__ var,
                     _Float16* __restrict__ out) {     // NHWC [B][64][64][64]
  constexpr bool NHWC_IN = (sizeof(IT) == 2);
  __shared__ __align__(32) _Float16 sIn[180 * 32];      // [y10][x18][cin32]
  __shared__ __align__(32) _Float16 sW9[9 * 64 * 32];   // [tap][N=64][K=32]

  const int tid  = threadIdx.x;
  const int wave = tid >> 5;
  const int lane = tid & 31;
  const int blk  = blockIdx.x;
  const int img  = blk >> 5;
  const int t    = blk & 31;
  const int y0   = (t >> 2) * 8;
  const int x0   = (t & 3) * 16;

  const int mcol  = lane & 15;  // pixel column within tile; also N within 16-group
  const int khalf = lane >> 4;  // which 16-element K half this lane holds

  v8f acc[4] = {};

  for (int c0 = 0; c0 < CIN; c0 += 32) {
    // --- stage ALL 9 taps of the weight tile, transposed to [tap][N][K] ---
    // global: one b128 (8 cout) per iteration, coalesced; LDS: 8 b16 transpose stores
    for (int i = tid; i < 9 * 256; i += 256) {
      int tap = i >> 8;
      int r   = i & 255;
      int k   = r >> 3;
      int n0  = (r & 7) << 3;
      v8h w = *(const v8h*)&wb[((size_t)tap * CIN + c0 + k) * 64 + n0];
      #pragma unroll
      for (int j = 0; j < 8; ++j) sW9[(tap * 64 + n0 + j) * 32 + k] = w[j];
    }

    // --- stage input halo [10][18][cin32] ---
    if constexpr (NHWC_IN) {
      // NHWC f16: 16B contiguous per (pixel, 8-channel group)
      for (int i = tid; i < 180 * 4; i += 256) {
        int pix = i >> 2, grp = i & 3;
        int yy = pix / 18, xx = pix % 18;
        int gy = y0 + yy - 1, gx = x0 + xx - 1;
        _Float16* dst = &sIn[pix * 32 + grp * 8];
        if (gy >= 0 && gy < 64 && gx >= 0 && gx < 64) {
          const _Float16* src =
              &in[(((size_t)img * 64 + gy) * 64 + gx) * CIN + c0 + grp * 8];
#if HAVE_ASYNC_LDS
          __builtin_amdgcn_global_load_async_to_lds_b128(
              (gbl_v4i*)src, (lds_v4i*)dst, 0, 0);
#else
          *(v8h*)dst = *(const v8h*)src;
#endif
        } else {
          v8h z = {};
          *(v8h*)dst = z;
        }
      }
#if HAVE_ASYNC_LDS
      __builtin_amdgcn_s_wait_asynccnt(0);
#endif
    } else {
      // NCHW f32: coalesced b32 loads along x, convert to f16 while staging
      if (c0 + 32 < CIN)
        __builtin_prefetch(&in[(((size_t)img * CIN + c0 + 32) * 64 + y0) * 64 + x0], 0, 1);
      for (int i = tid; i < 180 * 32; i += 256) {
        int c  = i / 180;
        int r  = i % 180;
        int yy = r / 18, xx = r % 18;
        int gy = y0 + yy - 1, gx = x0 + xx - 1;
        float v = 0.f;
        if (gy >= 0 && gy < 64 && gx >= 0 && gx < 64)
          v = (float)in[(((size_t)img * CIN + c0 + c) * 64 + gy) * 64 + gx];
        sIn[(yy * 18 + xx) * 32 + c] = (_Float16)v;
      }
    }
    __syncthreads();

    // --- 9 taps x 4 n-groups of WMMA, no barriers inside ---
    #pragma unroll
    for (int tap = 0; tap < 9; ++tap) {
      const int iy = wave + tap / 3;
      const int ix = mcol + tap % 3;
      const v16h a = *(const v16h*)&sIn[(iy * 18 + ix) * 32 + khalf * 16];
      #pragma unroll
      for (int ng = 0; ng < 4; ++ng) {
        const v16h bfr = *(const v16h*)&sW9[(tap * 64 + ng * 16 + mcol) * 32 + khalf * 16];
        acc[ng] = __builtin_amdgcn_wmma_f32_16x16x32_f16(
            false, a, false, bfr, (short)0, acc[ng], false, false);
      }
    }
    __syncthreads();
  }

  // Epilogue: BN (inference) + ReLU, store f16 activations NHWC.
  const int oy = y0 + wave;
  #pragma unroll
  for (int ng = 0; ng < 4; ++ng) {
    const int ch   = ng * 16 + mcol;
    const float sc = gam[ch] * rsqrtf(var[ch] + 1e-5f);
    const float bi = bet[ch] - mu[ch] * sc;
    #pragma unroll
    for (int r = 0; r < 8; ++r) {
      const int mrow = r + khalf * 8;  // pixel column per C/D VGPR layout
      float vv = acc[ng][r] * sc + bi;
      vv = vv > 0.f ? vv : 0.f;
      out[(((size_t)img * 64 + oy) * 64 + (x0 + mrow)) * 64 + ch] = (_Float16)vv;
    }
  }
}

// ------------------------------------------------------------------
// 1x1 conv over 64 channels + bias (NHWC: contiguous 64-half dot per pixel).
// ------------------------------------------------------------------
__global__ void conv1x1_bias(const _Float16* __restrict__ act,  // NHWC [B*4096][64]
                             const float* __restrict__ w3,
                             const float* __restrict__ b3,
                             float* __restrict__ out) {          // [B][4096]
  __shared__ float sw[64];
  if (threadIdx.x < 64) sw[threadIdx.x] = w3[threadIdx.x];
  __syncthreads();
  int idx = blockIdx.x * 256 + threadIdx.x;
  if (idx >= BATCH * 4096) return;
  const _Float16* a = act + (size_t)idx * 64;
  float s = b3[0];
  #pragma unroll
  for (int g = 0; g < 8; ++g) {
    v8h va = *(const v8h*)(a + g * 8);
    #pragma unroll
    for (int j = 0; j < 8; ++j) s += (float)va[j] * sw[g * 8 + j];
  }
  out[idx] = s;
}

// ------------------------------------------------------------------
// Half-pixel bilinear 64->256 (align_corners=False) + sigmoid.
// ------------------------------------------------------------------
__global__ void upsample_sigmoid(const float* __restrict__ l64,  // [B][64][64]
                                 float* __restrict__ heat,
                                 float* __restrict__ logits) {
  int idx = blockIdx.x * 256 + threadIdx.x;
  if (idx >= BATCH * OUTHW * OUTHW) return;
  int img = idx / (OUTHW * OUTHW);
  int p   = idx % (OUTHW * OUTHW);
  int oy = p >> 8, ox = p & 255;
  const float* src = l64 + (size_t)img * 4096;
  float fy = (oy + 0.5f) * 0.25f - 0.5f;
  float fx = (ox + 0.5f) * 0.25f - 0.5f;
  float y0f = floorf(fy), x0f = floorf(fx);
  float wy = fy - y0f, wx = fx - x0f;
  int iy0 = (int)y0f, ix0 = (int)x0f;
  int y0c = min(max(iy0, 0), 63),     y1c = min(max(iy0 + 1, 0), 63);
  int x0c = min(max(ix0, 0), 63),     x1c = min(max(ix0 + 1, 0), 63);
  float v00 = src[y0c * 64 + x0c], v01 = src[y0c * 64 + x1c];
  float v10 = src[y1c * 64 + x0c], v11 = src[y1c * 64 + x1c];
  float v = (1.f - wy) * ((1.f - wx) * v00 + wx * v01) +
            wy        * ((1.f - wx) * v10 + wx * v11);
  logits[idx] = v;
  heat[idx]   = 1.f / (1.f + __expf(-v));
}

// ------------------------------------------------------------------
// Per-image point sampling: top-3 9x9-NMS peaks (>0.1) + argmax fallback,
// 2 "uniform" negatives from low-response region. One block per image.
// ------------------------------------------------------------------
__global__ __launch_bounds__(256)
void sample_points(const float* __restrict__ heat,  // [B][256][256]
                   float* __restrict__ coords,      // [B][1][5][2]
                   int*   __restrict__ labels) {    // [B][1][5]
  const int img = blockIdx.x;
  const int tid = threadIdx.x;
  const float* h = heat + (size_t)img * 65536;
  __shared__ float rv[256];
  __shared__ int   ri[256];
  __shared__ int   sel[3];
  __shared__ int   amaxi;
  __shared__ float amaxv;
  __shared__ int   lowcnt;

  // global argmax
  float bv = -1.f; int bi = 0;
  for (int p = tid; p < 65536; p += 256) { float x = h[p]; if (x > bv) { bv = x; bi = p; } }
  rv[tid] = bv; ri[tid] = bi; __syncthreads();
  for (int s = 128; s > 0; s >>= 1) {
    if (tid < s && rv[tid + s] > rv[tid]) { rv[tid] = rv[tid + s]; ri[tid] = ri[tid + s]; }
    __syncthreads();
  }
  if (tid == 0) { amaxi = ri[0]; amaxv = rv[0]; sel[0] = sel[1] = sel[2] = -1; }
  __syncthreads();

  // top-3 NMS peaks via repeated masked argmax
  for (int slot = 0; slot < 3; ++slot) {
    float lv = -1.f; int li = -1;
    for (int p = tid; p < 65536; p += 256) {
      if (p == sel[0] || p == sel[1] || p == sel[2]) continue;
      float x = h[p];
      if (x <= 0.1f || x <= lv) continue;
      int y = p >> 8, xx = p & 255;
      bool peak = true;
      for (int dy = -4; dy <= 4 && peak; ++dy) {
        int yy = y + dy; if (yy < 0 || yy > 255) continue;
        for (int dx = -4; dx <= 4; ++dx) {
          int xc = xx + dx; if (xc < 0 || xc > 255) continue;
          if (h[yy * 256 + xc] > x) { peak = false; break; }
        }
      }
      if (peak) { lv = x; li = p; }
    }
    rv[tid] = lv; ri[tid] = li; __syncthreads();
    for (int s = 128; s > 0; s >>= 1) {
      if (tid < s && rv[tid + s] > rv[tid]) { rv[tid] = rv[tid + s]; ri[tid] = ri[tid + s]; }
      __syncthreads();
    }
    if (tid == 0) sel[slot] = (rv[0] > 0.1f) ? ri[0] : -1;
    __syncthreads();
  }

  // count low-response pixels
  int cnt = 0;
  for (int p = tid; p < 65536; p += 256) if (h[p] < 0.3f) ++cnt;
  ri[tid] = cnt; __syncthreads();
  for (int s = 128; s > 0; s >>= 1) { if (tid < s) ri[tid] += ri[tid + s]; __syncthreads(); }
  if (tid == 0) lowcnt = ri[0];
  __syncthreads();

  if (tid == 0) {
    float* C = coords + img * 10;
    int*   L = labels + img * 5;
    for (int s = 0; s < 3; ++s) {
      int p = sel[s];
      bool valid = (p >= 0);
      float x = valid ? (float)(p & 255) : 0.f;
      float y = valid ? (float)(p >> 8)  : 0.f;
      int lab = valid ? 1 : -1;
      if (s == 0) {
        if (!valid) { x = (float)(amaxi & 255); y = (float)(amaxi >> 8); }
        lab = 1;
      }
      C[s * 2 + 0] = x; C[s * 2 + 1] = y; L[s] = lab;
    }
    int n = lowcnt;
    float thresh = (n > 0) ? 0.3f : amaxv;   // fallback mask: h < max(h)
    if (n <= 0) { n = 0; for (int p = 0; p < 65536; ++p) if (h[p] < thresh) ++n; }
    int ridx[2];
    for (int j = 0; j < 2; ++j) {            // deterministic SplitMix64 stand-in PRNG
      unsigned long long z = 42ull * 0x9E3779B97F4A7C15ull +
                             (unsigned long long)(img * 2 + j + 1) * 0xBF58476D1CE4E5B9ull;
      z ^= z >> 30; z *= 0xBF58476D1CE4E5B9ull;
      z ^= z >> 27; z *= 0x94D049BB133111EBull; z ^= z >> 31;
      float u = (float)((z >> 40) & 0xFFFFFF) * (1.0f / 16777216.0f);
      int nn = n > 1 ? n : 1;
      int r = (int)(u * (float)nn);
      int cap = n - 1; if (cap < 0) cap = 0;
      ridx[j] = r < cap ? r : cap;
    }
    int selN[2] = {0, 0};
    int c2 = 0;
    for (int p = 0; p < 65536; ++p) {
      if (h[p] < thresh) {
        if (c2 == ridx[0]) selN[0] = p;
        if (c2 == ridx[1]) selN[1] = p;
        ++c2;
      }
    }
    for (int j = 0; j < 2; ++j) {
      bool any = (n > 0);
      C[6 + j * 2 + 0] = any ? (float)(selN[j] & 255) : 0.f;
      C[6 + j * 2 + 1] = any ? (float)(selN[j] >> 8)  : 0.f;
      L[3 + j] = 0;
    }
  }
}

// ------------------------------------------------------------------
extern "C" void kernel_launch(void* const* d_in, const int* in_sizes, int n_in,
                              void* d_out, int out_size, void* d_ws, size_t ws_size,
                              hipStream_t stream) {
  const float* x  = (const float*)d_in[0];
  const float* w1 = (const float*)d_in[1];
  const float* g1 = (const float*)d_in[2];
  const float* b1 = (const float*)d_in[3];
  const float* m1 = (const float*)d_in[4];
  const float* v1 = (const float*)d_in[5];
  const float* w2 = (const float*)d_in[6];
  const float* g2 = (const float*)d_in[7];
  const float* b2 = (const float*)d_in[8];
  const float* m2 = (const float*)d_in[9];
  const float* v2 = (const float*)d_in[10];
  const float* w3 = (const float*)d_in[11];
  const float* b3 = (const float*)d_in[12];

  // workspace layout (bytes)
  char* ws = (char*)d_ws;
  _Float16* wb1  = (_Float16*)(ws);                             // 294912 B
  _Float16* wb2  = (_Float16*)(ws + 294912);                    //  73728 B
  _Float16* act1 = (_Float16*)(ws + 368640);                    // 16 MiB (NHWC)
  _Float16* act2 = (_Float16*)(ws + 368640 + 16777216);         // 16 MiB (NHWC)
  float*    l64  = (float*)   (ws + 368640 + 2 * 16777216);     // 512 KiB

  // output layout (flat concat, reference return order)
  float* out    = (float*)d_out;
  float* heat   = out;                         // 32*1*256*256
  float* coords = out + 2097152;               // 32*1*5*2
  int*   labels = (int*)(out + 2097472);       // 32*1*5 (int32 bits)
  float* logits = out + 2097632;               // 32*1*256*256

  repack_weights<<<(64 * 256 * 9 + 64 * 64 * 9 + 255) / 256, 256, 0, stream>>>(w1, w2, wb1, wb2);
  conv3x3_bn_relu<256, float><<<1024, 256, 0, stream>>>(x, wb1, g1, b1, m1, v1, act1);
  conv3x3_bn_relu<64, _Float16><<<1024, 256, 0, stream>>>(act1, wb2, g2, b2, m2, v2, act2);
  conv1x1_bias<<<(BATCH * 4096 + 255) / 256, 256, 0, stream>>>(act2, w3, b3, l64);
  upsample_sigmoid<<<(BATCH * OUTHW * OUTHW + 255) / 256, 256, 0, stream>>>(l64, heat, logits);
  sample_points<<<BATCH, 256, 0, stream>>>(heat, coords, labels);
}
